// _SelfAttentionBlock_257698038077
// MI455X (gfx1250) — matrix operand order repaired and checked by reading, not compiled, not run on hardware
//
#include <hip/hip_runtime.h>

// Problem constants (from reference): x[N,C,H,W], S=H*W
#define NB   2
#define CIN  512
#define SS   9216   // 96*96
#define KCC  256    // key/query channels
#define VCC  256    // value channels
#define COO  512    // output channels

typedef __attribute__((ext_vector_type(16))) __bf16 v16bf;
typedef __attribute__((ext_vector_type(8)))  __bf16 v8bf;
typedef __attribute__((ext_vector_type(8)))  float  v8f;

// ---------------------------------------------------------------------------
// WMMA fragment loaders (CDNA5 wave32 layouts, CDNA5 ISA §7.12.2)
//
// A 16x32 bf16 (MxK): lane L -> row m = L&15, half h = L>>4.
//   elements 0..7  hold K = h*8 .. h*8+7       (contiguous 16B)
//   elements 8..15 hold K = 16+h*8 .. 16+h*8+7 (contiguous 16B)
// Source matrix row-major [row][K], p pre-offset to the K-chunk base.
static __device__ __forceinline__ v16bf load_a_frag(const __bf16* p, int ld,
                                                    int row, int h) {
  union { v16bf v; v8bf x[2]; } f;
  const __bf16* b = p + (size_t)row * ld + h * 8;
  f.x[0] = *(const v8bf*)(b);
  f.x[1] = *(const v8bf*)(b + 16);
  return f.v;
}

// B 32x16 bf16 (KxN): lane L -> col n = L&15, half h = L>>4,
//   elements 0..7 hold K = h*8 .. h*8+7, elements 8..15 hold K = 16+h*8 .. 16+h*8+7.
// Source matrix stored so each output-column's K-vector is contiguous:
// layout [col][K] row-major, p pre-offset to the K-chunk base.
static __device__ __forceinline__ v16bf load_b_frag(const __bf16* p, int ld,
                                                    int col, int h) {
  union { v16bf v; v8bf x[2]; } f;
  const __bf16* b = p + (size_t)col * ld + h * 8;
  f.x[0] = *(const v8bf*)(b);
  f.x[1] = *(const v8bf*)(b + 16);
  return f.v;
}

// ---------------------------------------------------------------------------
// 1) Fold BN into Wk, convert all weights to bf16, fold key bias.
__global__ void __launch_bounds__(256)
sa_prep_weights(const float* __restrict__ Wk, const float* __restrict__ bk,
                const float* __restrict__ gamma, const float* __restrict__ beta,
                const float* __restrict__ rmean, const float* __restrict__ rvar,
                const float* __restrict__ Wv, const float* __restrict__ Ww,
                __bf16* __restrict__ Wkb, float* __restrict__ bkf,
                __bf16* __restrict__ Wvb, __bf16* __restrict__ Wwb) {
  int tid = blockIdx.x * blockDim.x + threadIdx.x;
  int np  = gridDim.x * blockDim.x;
  for (int i = tid; i < KCC * CIN; i += np) {
    int kc = i / CIN;
    float s = gamma[kc] * rsqrtf(rvar[kc] + 1e-5f);
    Wkb[i] = (__bf16)(Wk[i] * s);
  }
  for (int i = tid; i < VCC * CIN; i += np) Wvb[i] = (__bf16)Wv[i];
  for (int i = tid; i < COO * VCC; i += np) Wwb[i] = (__bf16)Ww[i];
  for (int i = tid; i < KCC; i += np) {
    float s = gamma[i] * rsqrtf(rvar[i] + 1e-5f);
    bkf[i] = bk[i] * s + beta[i] - rmean[i] * s;
  }
}

// ---------------------------------------------------------------------------
// 2) Transpose x[N][C][S] f32 -> xT[N][S][C] bf16 via LDS 32x32 tile.
__global__ void __launch_bounds__(256)
sa_transpose_x(const float* __restrict__ x, __bf16* __restrict__ xT) {
  __shared__ __bf16 tile[32][33];
  int n = blockIdx.z;
  int c0 = blockIdx.y * 32, s0 = blockIdx.x * 32;
  int tx = threadIdx.x, ty = threadIdx.y;
  const float* xp = x + ((size_t)n * CIN + c0) * SS + s0;
  for (int i = ty; i < 32; i += 8)
    tile[i][tx] = (__bf16)xp[(size_t)i * SS + tx];     // tile[c][s]
  __syncthreads();
  __bf16* op = xT + ((size_t)n * SS + s0) * CIN + c0;
  for (int i = ty; i < 32; i += 8)
    op[(size_t)i * CIN + tx] = tile[tx][i];            // xT[s][c] = x[c][s]
}

// ---------------------------------------------------------------------------
// 3) Generic WMMA projection: P[R x S] = W[R x kdim] * B[S x kdim]^T + bias
//    MODE 0: bf16 out, token-major [S][KCC], ReLU   (key/query path, BN folded)
//    MODE 1: bf16 out, channel-major [R][S]         (value path)
//    MODE 2: f32  out, channel-major [R][S]         (final output projection)
// Block = 4 waves; block tile = 64 rows x 64 cols; wave owns 16 rows.
template <int MODE>
__global__ void __launch_bounds__(128, 1)
sa_proj(const __bf16* __restrict__ W, const float* __restrict__ bias,
        const __bf16* __restrict__ Bmat, int kdim, int R,
        __bf16* __restrict__ outb, float* __restrict__ outf) {
  int n    = blockIdx.z;
  int r0   = blockIdx.y * 64, s0 = blockIdx.x * 64;
  int wave = threadIdx.x >> 5, lane = threadIdx.x & 31;
  int m    = lane & 15, h = lane >> 4;
  int rw   = r0 + wave * 16;
  const __bf16* Bb = Bmat + ((size_t)n * SS + s0) * kdim;

  v8f acc[4] = {};
  for (int cb = 0; cb < kdim; cb += 32) {
    v16bf a = load_a_frag(W + cb, kdim, rw + m, h);
#pragma unroll
    for (int t = 0; t < 4; ++t) {
      v16bf b = load_b_frag(Bb + cb, kdim, t * 16 + m, h);
      acc[t] = __builtin_amdgcn_wmma_f32_16x16x32_bf16(false, a, false, b,
                                                       (short)0, acc[t], false, false);
    }
  }
  // D fragment: lane covers col n = L&15; vgpr r covers row M = r + 8*h.
  float bb[8];
#pragma unroll
  for (int r = 0; r < 8; ++r) bb[r] = bias[rw + r + 8 * h];
#pragma unroll
  for (int t = 0; t < 4; ++t) {
#pragma unroll
    for (int r = 0; r < 8; ++r) {
      float v  = acc[t][r] + bb[r];
      int row  = rw + r + 8 * h;
      int col  = s0 + t * 16 + m;
      if (MODE == 0)      outb[((size_t)n * SS + col) * KCC + row] = (__bf16)fmaxf(v, 0.f);
      else if (MODE == 1) outb[((size_t)n * R  + row) * SS  + col] = (__bf16)v;
      else                outf[((size_t)n * R  + row) * SS  + col] = v;
    }
  }
}

// ---------------------------------------------------------------------------
// 4) Flash attention over S=9216 tokens. Q==K (shared transform).
//    Block = 4 waves; each wave owns 16 query rows (block: 64 rows).
//    Streams keys 32 at a time with online softmax; ctx accumulated in f32.
//    launch_bounds(128,1): ~240 live VGPRs/lane (128 acc + 64 Q frags + temps)
//    must stay resident — default occupancy targets caused scratch spills.
__global__ void __launch_bounds__(128, 1)
sa_attn(const __bf16* __restrict__ Kq,   // [N][S][KCC]
        const __bf16* __restrict__ Vt,   // [N][VCC][S]
        __bf16* __restrict__ Ct) {       // [N][S][VCC]
  int n    = blockIdx.y;
  int q0   = blockIdx.x * 64;
  int wave = threadIdx.x >> 5, lane = threadIdx.x & 31;
  int m    = lane & 15, h = lane >> 4;
  int qw   = q0 + wave * 16;

  const __bf16* Kb = Kq + (size_t)n * SS * KCC;
  const __bf16* Vb = Vt + (size_t)n * VCC * SS;

  // Wave-private LDS (no cross-wave sharing -> no __syncthreads in the loop;
  // same-wave DS ordering via DScnt guarantees visibility).
  __shared__ __align__(32) float  sS[4][16][36];   // scores, padded stride
  __shared__ __align__(32) __bf16 sP[4][16][32];   // exp(probabilities)
  __shared__ float sM[4][16], sL[4][16], sF[4][16];

  // Q fragments for this wave's 16 rows: KCC=256 -> 8 chunks of K=32.
  v16bf qf[8];
#pragma unroll
  for (int c = 0; c < 8; ++c)
    qf[c] = load_a_frag(Kb + c * 32, KCC, qw + m, h);

  v8f acc[16] = {};                                 // ctx accum: 16 tiles over VCC
  if (lane < 16) { sM[wave][lane] = -3.0e38f; sL[wave][lane] = 0.f; }

  for (int jb = 0; jb < SS; jb += 32) {
    if (jb + 32 < SS)
      __builtin_prefetch(Kb + (size_t)(jb + 32) * KCC, 0, 1);  // global_prefetch_b8

    // --- scores: sim[16 q x 32 keys] = Q * K^T, scaled by 1/sqrt(256)
#pragma unroll
    for (int kt = 0; kt < 2; ++kt) {
      v8f sacc = {};
#pragma unroll
      for (int c = 0; c < 8; ++c) {
        v16bf kf = load_b_frag(Kb + (size_t)(jb + kt * 16) * KCC + c * 32, KCC, m, h);
        sacc = __builtin_amdgcn_wmma_f32_16x16x32_bf16(false, qf[c], false, kf,
                                                       (short)0, sacc, false, false);
      }
#pragma unroll
      for (int r = 0; r < 8; ++r)
        sS[wave][r + 8 * h][kt * 16 + m] = sacc[r] * 0.0625f;
    }

    // --- online softmax: lane j (<16) owns query row j of this wave
    if (lane < 16) {
      float mo = sM[wave][lane];
      float mx = mo;
#pragma unroll
      for (int j = 0; j < 32; ++j) mx = fmaxf(mx, sS[wave][lane][j]);
      float f = __expf(mo - mx);
      float l = sL[wave][lane] * f;
#pragma unroll
      for (int j = 0; j < 32; ++j) {
        float p = __expf(sS[wave][lane][j] - mx);
        l += p;
        sP[wave][lane][j] = (__bf16)p;
      }
      sM[wave][lane] = mx; sL[wave][lane] = l; sF[wave][lane] = f;
    }

    // --- rescale running ctx accumulators by exp(m_old - m_new) per row
    float fr[8];
#pragma unroll
    for (int r = 0; r < 8; ++r) fr[r] = sF[wave][r + 8 * h];
#pragma unroll
    for (int t = 0; t < 16; ++t)
#pragma unroll
      for (int r = 0; r < 8; ++r) acc[t][r] *= fr[r];

    // --- P as A-fragment (16 q-rows x 32 keys) from LDS
    union { v16bf v; v8bf x[2]; } pf;
    pf.x[0] = *(const v8bf*)(&sP[wave][m][h * 8]);
    pf.x[1] = *(const v8bf*)(&sP[wave][m][16 + h * 8]);

    // --- ctx += P * V : 16 tiles across VCC=256, V loads pipelined ahead
    v16bf vf = load_b_frag(Vb + (size_t)0 * SS + jb, SS, m, h);
#pragma unroll
    for (int t = 0; t < 16; ++t) {
      v16bf vfn;
      if (t < 15) vfn = load_b_frag(Vb + (size_t)((t + 1) * 16) * SS + jb, SS, m, h);
      acc[t] = __builtin_amdgcn_wmma_f32_16x16x32_bf16(false, pf.v, false, vf,
                                                       (short)0, acc[t], false, false);
      vf = vfn;
    }
  }

  // --- finalize: divide by row sums, store ctx token-major [S][VCC] bf16
  float li[8];
#pragma unroll
  for (int r = 0; r < 8; ++r) li[r] = 1.0f / sL[wave][r + 8 * h];
  __bf16* Cb = Ct + (size_t)n * SS * VCC;
#pragma unroll
  for (int t = 0; t < 16; ++t)
#pragma unroll
    for (int r = 0; r < 8; ++r)
      Cb[(size_t)(qw + r + 8 * h) * VCC + t * 16 + m] = (__bf16)(acc[t][r] * li[r]);
}

// ---------------------------------------------------------------------------
extern "C" void kernel_launch(void* const* d_in, const int* in_sizes, int n_in,
                              void* d_out, int out_size, void* d_ws, size_t ws_size,
                              hipStream_t stream) {
  const float* x     = (const float*)d_in[0];
  const float* Wk    = (const float*)d_in[1];
  const float* bk    = (const float*)d_in[2];
  const float* gamma = (const float*)d_in[3];
  const float* beta  = (const float*)d_in[4];
  const float* rmean = (const float*)d_in[5];
  const float* rvar  = (const float*)d_in[6];
  const float* Wv    = (const float*)d_in[7];
  const float* bv    = (const float*)d_in[8];
  const float* Ww    = (const float*)d_in[9];
  const float* bw    = (const float*)d_in[10];
  float* out = (float*)d_out;

  char* ws = (char*)d_ws;
  size_t off = 0;
  auto alloc = [&](size_t bytes) -> void* {
    void* p = ws + off;
    off = (off + bytes + 255) & ~(size_t)255;
    return p;
  };
  __bf16* xT  = (__bf16*)alloc((size_t)NB * SS * CIN * 2);  // 18.9 MB (reused as Ct)
  __bf16* Kq  = (__bf16*)alloc((size_t)NB * SS * KCC * 2);  // 9.44 MB
  __bf16* Vt  = (__bf16*)alloc((size_t)NB * VCC * SS * 2);  // 9.44 MB
  __bf16* Wkb = (__bf16*)alloc((size_t)KCC * CIN * 2);
  __bf16* Wvb = (__bf16*)alloc((size_t)VCC * CIN * 2);
  __bf16* Wwb = (__bf16*)alloc((size_t)COO * VCC * 2);
  float*  bkf = (float*)alloc((size_t)KCC * 4);
  __bf16* Ct  = xT;  // xT is dead after the K/V projections; reuse for ctx

  sa_prep_weights<<<64, 256, 0, stream>>>(Wk, bk, gamma, beta, rmean, rvar,
                                          Wv, Ww, Wkb, bkf, Wvb, Wwb);
  sa_transpose_x<<<dim3(SS / 32, CIN / 32, NB), dim3(32, 8), 0, stream>>>(x, xT);
  sa_proj<0><<<dim3(SS / 64, KCC / 64, NB), 128, 0, stream>>>(Wkb, bkf, xT, CIN, KCC, Kq, nullptr);
  sa_proj<1><<<dim3(SS / 64, VCC / 64, NB), 128, 0, stream>>>(Wvb, bv,  xT, CIN, VCC, Vt, nullptr);
  sa_attn<<<dim3(SS / 64, NB), 128, 0, stream>>>(Kq, Vt, Ct);
  sa_proj<2><<<dim3(SS / 64, COO / 64, NB), 128, 0, stream>>>(Wwb, bw, Ct, VCC, COO, nullptr, out);
}
